// MultiHead_45784351375719
// MI455X (gfx1250) — matrix operand order repaired and checked by reading, not compiled, not run on hardware
//
#include <hip/hip_runtime.h>

typedef __attribute__((ext_vector_type(16))) _Float16 v16h;
typedef __attribute__((ext_vector_type(8)))  float    v8f;

namespace {
constexpr int   kS     = 11;   // sequence length
constexpr int   kD     = 32;   // model dim
constexpr int   kH     = 4;    // heads
constexpr int   kDH    = 8;    // head dim
constexpr int   kRow   = 32;   // LDS row stride (floats)
constexpr int   kWaves = 4;    // waves per block
constexpr float kEps   = 1e-5f;
constexpr float kScale = 0.35355339059327373f; // 1/sqrt(8)
}

__device__ __forceinline__ v8f wmma_f16(v16h a, v16h b, v8f c) {
  // D = A(16x32 f16) * B(32x16 f16) + C(16x16 f32)
  return __builtin_amdgcn_wmma_f32_16x16x32_f16(
      /*neg_a=*/false, a, /*neg_b=*/false, b,
      /*c_mod=*/(short)0, c, /*reuse_a=*/false, /*reuse_b=*/false);
}

// B-matrix fragment for y = x @ W^T : B[k][n] = W[n][k].
// 32x16 f16 B layout (wave32): lane (n = lane&15 [+16 for tile1]) holds
// h[0..7]=K 8*(lane>>4)+.., h[8..15]=K 16+8*(lane>>4)+.. (matches A).
__device__ __forceinline__ v16h make_bfrag(const float* __restrict__ W, int n, int k0) {
  v16h h;
#pragma unroll
  for (int i = 0; i < 16; ++i) h[i] = (_Float16)W[n * kD + (i < 8 ? (k0 >> 1) + i : 16 + (k0 >> 1) + (i - 8))];
  return h;
}

__global__ __launch_bounds__(kWaves * 32)
void mha_ln_fused(const float* __restrict__ Q, const float* __restrict__ K,
                  const float* __restrict__ V, const unsigned char* __restrict__ mask,
                  const float* __restrict__ Wq, const float* __restrict__ bq,
                  const float* __restrict__ Wk, const float* __restrict__ bk,
                  const float* __restrict__ Wv, const float* __restrict__ bv,
                  const float* __restrict__ Wo, const float* __restrict__ bo,
                  const float* __restrict__ gamma, const float* __restrict__ beta,
                  float* __restrict__ out, int nBatch)
{
  // per-wave scratch: 0=xQ(residual,f32) 1=q 2=k 3=v 4=attn   (16x32 f32 each)
  __shared__ float lds[kWaves][5][16 * kRow];

  const int lane = threadIdx.x & 31;
  const int wib  = threadIdx.x >> 5;
  const int hi   = lane >> 4;     // K-half selector for A/B fragments
  const int lo   = lane & 15;
  const int mrow = lo;            // A-fragment row owned by this lane
  const int n0   = lo;            // C/D column, tile 0
  const int n1   = lo + 16;       // C/D column, tile 1
  const int k0   = 16 * hi;       // B-fragment K base

  float* xq_s = lds[wib][0];
  float* q_s  = lds[wib][1];
  float* k_s  = lds[wib][2];
  float* v_s  = lds[wib][3];
  float* at_s = lds[wib][4];

  // Stationary weight fragments (L2-resident 4KB, read once per wave)
  v16h wqf0 = make_bfrag(Wq, n0, k0), wqf1 = make_bfrag(Wq, n1, k0);
  v16h wkf0 = make_bfrag(Wk, n0, k0), wkf1 = make_bfrag(Wk, n1, k0);
  v16h wvf0 = make_bfrag(Wv, n0, k0), wvf1 = make_bfrag(Wv, n1, k0);
  v16h wof0 = make_bfrag(Wo, n0, k0), wof1 = make_bfrag(Wo, n1, k0);

  const float bq0 = bq[n0], bq1 = bq[n1];
  const float bk0 = bk[n0], bk1 = bk[n1];
  const float bv0 = bv[n0], bv1 = bv[n1];
  const float bo0 = bo[n0], bo1 = bo[n1];
  const float g0  = gamma[n0], g1 = gamma[n1];
  const float be0 = beta[n0],  be1 = beta[n1];

  const long long gw = (long long)blockIdx.x * kWaves + wib;
  const long long nw = (long long)gridDim.x * kWaves;
  const bool rowok = (mrow < kS);

  for (long long b = gw; b < nBatch; b += nw) {
    const float* Qb = Q + b * (kS * kD);
    const float* Kb = K + b * (kS * kD);
    const float* Vb = V + b * (kS * kD);
    const unsigned char* Mb = mask + b * (kS * kS);

    if (b + nw < nBatch) {  // pull next batch toward the caches
      const long long off = nw * (kS * kD) + (long long)lane * (kS * kD / 32);
      __builtin_prefetch(Qb + off, 0, 0);
      __builtin_prefetch(Kb + off, 0, 0);
      __builtin_prefetch(Vb + off, 0, 0);
    }

    // ---- Build A fragments (f16) straight from global; stash f32 Q for residual.
    // 16-bit A 16x32 layout: lane holds row mrow; h[0..7]=x[m][8*hi+..],
    // h[8..15]=x[m][16+8*hi+..]. Rows >= S are zero (also guards OOB).
    v16h aq, ak, av;
#pragma unroll
    for (int i = 0; i < 16; ++i) {
      aq[i] = (_Float16)0.f; ak[i] = (_Float16)0.f; av[i] = (_Float16)0.f;
    }
    if (rowok) {
#pragma unroll
      for (int i = 0; i < 8; ++i) {
        float x0 = Qb[mrow * kD + 8 * hi + i];
        float x1 = Qb[mrow * kD + 16 + 8 * hi + i];
        aq[i] = (_Float16)x0; aq[8 + i] = (_Float16)x1;
        xq_s[mrow * kRow + 8 * hi + i]      = x0;
        xq_s[mrow * kRow + 16 + 8 * hi + i] = x1;
        ak[i]     = (_Float16)Kb[mrow * kD + 8 * hi + i];
        ak[8 + i] = (_Float16)Kb[mrow * kD + 16 + 8 * hi + i];
        av[i]     = (_Float16)Vb[mrow * kD + 8 * hi + i];
        av[8 + i] = (_Float16)Vb[mrow * kD + 16 + 8 * hi + i];
      }
    } else {
#pragma unroll
      for (int i = 0; i < 8; ++i) {
        xq_s[mrow * kRow + 8 * hi + i]      = 0.f;
        xq_s[mrow * kRow + 16 + 8 * hi + i] = 0.f;
      }
    }

    // ---- QKV projections: 2 WMMA each (K=32 in one op, two N-tiles).
    // C/D layout: VGPR j -> row (j + 8*hi), column lane&15 (+16 for tile1).
    v8f c0 = {}, c1 = {};
    c0 = wmma_f16(aq, wqf0, c0);
    c1 = wmma_f16(aq, wqf1, c1);
#pragma unroll
    for (int j = 0; j < 8; ++j) {
      int mm = j + 8 * hi;
      q_s[mm * kRow + n0] = c0[j] + bq0;
      q_s[mm * kRow + n1] = c1[j] + bq1;
    }
    v8f d0 = {}, d1 = {};
    d0 = wmma_f16(ak, wkf0, d0);
    d1 = wmma_f16(ak, wkf1, d1);
#pragma unroll
    for (int j = 0; j < 8; ++j) {
      int mm = j + 8 * hi;
      k_s[mm * kRow + n0] = d0[j] + bk0;
      k_s[mm * kRow + n1] = d1[j] + bk1;
    }
    v8f e0 = {}, e1 = {};
    e0 = wmma_f16(av, wvf0, e0);
    e1 = wmma_f16(av, wvf1, e1);
#pragma unroll
    for (int j = 0; j < 8; ++j) {
      int mm = j + 8 * hi;
      v_s[mm * kRow + n0] = e0[j] + bv0;
      v_s[mm * kRow + n1] = e1[j] + bv1;
    }

    // Zero attn pad rows (S..15) so the output-projection A fragment is clean.
#pragma unroll
    for (int idx = lane; idx < (16 - kS) * kRow; idx += 32)
      at_s[kS * kRow + idx] = 0.f;

    asm volatile("" ::: "memory");  // wave-private LDS: DS is in-order per wave

    // ---- Attention core in VALU: one lane per (head, query-row); 44 rows -> 2 passes.
    for (int pass = 0; pass < 2; ++pass) {
      int r = pass * 32 + lane;
      if (r < kH * kS) {
        int h  = r / kS;
        int mq = r - h * kS;
        const int co = h * kDH;
        float qr[kDH];
#pragma unroll
        for (int d = 0; d < kDH; ++d) qr[d] = q_s[mq * kRow + co + d];
        float p[kS];
        float mx = -3.402823466e38f;
#pragma unroll
        for (int j = 0; j < kS; ++j) {
          float s = 0.f;
#pragma unroll
          for (int d = 0; d < kDH; ++d) s += qr[d] * k_s[j * kRow + co + d];
          s *= kScale;
          if (Mb[mq * kS + j] != 0) s = -__builtin_inff();
          p[j] = s;
          mx = fmaxf(mx, s);
        }
        float sum = 0.f;
#pragma unroll
        for (int j = 0; j < kS; ++j) { float e = __expf(p[j] - mx); p[j] = e; sum += e; }
        float inv = 1.f / sum;
        float acc[kDH];
#pragma unroll
        for (int d = 0; d < kDH; ++d) acc[d] = 0.f;
#pragma unroll
        for (int j = 0; j < kS; ++j)
#pragma unroll
          for (int d = 0; d < kDH; ++d) acc[d] += p[j] * v_s[j * kRow + co + d];
#pragma unroll
        for (int d = 0; d < kDH; ++d) at_s[mq * kRow + co + d] = acc[d] * inv;
      }
    }

    asm volatile("" ::: "memory");

    // ---- Output projection: 2 WMMA, then bias + residual + layernorm + store.
    v16h aa;
#pragma unroll
    for (int i = 0; i < 8; ++i) {
      aa[i]     = (_Float16)at_s[mrow * kRow + 8 * hi + i];
      aa[8 + i] = (_Float16)at_s[mrow * kRow + 16 + 8 * hi + i];
    }
    v8f o0 = {}, o1 = {};
    o0 = wmma_f16(aa, wof0, o0);
    o1 = wmma_f16(aa, wof1, o1);

#pragma unroll
    for (int j = 0; j < 8; ++j) {
      int mm = j + 8 * hi;
      float x0 = o0[j] + bo0 + xq_s[mm * kRow + n0];
      float x1 = o1[j] + bo1 + xq_s[mm * kRow + n1];
      float s  = x0 + x1;
      float ss = x0 * x0 + x1 * x1;
#pragma unroll
      for (int off = 1; off < 16; off <<= 1) {   // stays within each 16-lane half
        s  += __shfl_xor(s,  off, 32);
        ss += __shfl_xor(ss, off, 32);
      }
      float mu  = s * (1.f / 32.f);
      float var = ss * (1.f / 32.f) - mu * mu;   // population variance (jnp.var)
      float rin = rsqrtf(var + kEps);
      if (mm < kS) {
        float* op = out + (b * kS + mm) * kD;
        op[n0] = (x0 - mu) * rin * g0 + be0;     // lanes 0..15 contiguous: coalesced
        op[n1] = (x1 - mu) * rin * g1 + be1;
      }
    }

    asm volatile("" ::: "memory");
  }
}

extern "C" void kernel_launch(void* const* d_in, const int* in_sizes, int n_in,
                              void* d_out, int out_size, void* d_ws, size_t ws_size,
                              hipStream_t stream) {
  (void)n_in; (void)out_size; (void)d_ws; (void)ws_size;
  const float* Q  = (const float*)d_in[0];
  const float* K  = (const float*)d_in[1];
  const float* V  = (const float*)d_in[2];
  const unsigned char* mask = (const unsigned char*)d_in[3];
  const float* Wq = (const float*)d_in[4];
  const float* bq = (const float*)d_in[5];
  const float* Wk = (const float*)d_in[6];
  const float* bk = (const float*)d_in[7];
  const float* Wv = (const float*)d_in[8];
  const float* bv = (const float*)d_in[9];
  const float* Wo = (const float*)d_in[10];
  const float* bo = (const float*)d_in[11];
  const float* gamma = (const float*)d_in[12];
  const float* beta  = (const float*)d_in[13];

  const int B = in_sizes[0] / (kS * kD);
  int blocks = (B + kWaves - 1) / kWaves;
  if (blocks > 4096) blocks = 4096;          // 16384 waves, grid-stride over batches
  if (blocks < 1) blocks = 1;

  mha_ln_fused<<<dim3(blocks), dim3(kWaves * 32), 0, stream>>>(
      Q, K, V, mask, Wq, bq, Wk, bk, Wv, bv, Wo, bo, gamma, beta,
      (float*)d_out, B);
}
